// SnakeHead_56667798503563
// MI455X (gfx1250) — matrix operand order repaired and checked by reading, not compiled, not run on hardware
//
#include <hip/hip_runtime.h>

typedef __attribute__((ext_vector_type(16))) _Float16 v16h;
typedef __attribute__((ext_vector_type(8)))  _Float16 v8h;
typedef __attribute__((ext_vector_type(8)))  float    v8f;
typedef __attribute__((ext_vector_type(4)))  float    v4f;

// GCC-style int4 + address-space-qualified pointers for the async-to-LDS builtin
typedef int v4i __attribute__((vector_size(16)));
typedef __attribute__((address_space(1))) v4i* as1_v4i_p;
typedef __attribute__((address_space(3))) v4i* as3_v4i_p;

#define B_    8
#define N_    1024
#define H_    256
#define W_    256
#define C_    256
#define HID_  512
#define MAXR  9
#define SLAB_ROWS   50   // 32 rows + 2*MAXR halo
#define SLAB_STRIDE 40   // halfs per row: 32 + 8 pad (80B) -> conflict-free b128 reads

#if defined(__has_builtin)
#if __has_builtin(__builtin_amdgcn_global_load_async_to_lds_b128) && \
    __has_builtin(__builtin_amdgcn_s_wait_asynccnt)
#define USE_ASYNC 1
#endif
#endif
#ifndef USE_ASYNC
#define USE_ASYNC 0
#endif

// ---------------------------------------------------------------------------
// Pack fp32 conv weights [3, Cin, 512] into WMMA B-fragment order (f16):
// frag index (k, cin-chunk c, co-frag f); within frag: [lane 0..31][half j 0..15]
// value = W[k][c*32 + (lane>>4)*16 + j][f*16 + (lane&15)]
// ---------------------------------------------------------------------------
__global__ void pack_weights(const float* __restrict__ w, _Float16* __restrict__ p, int cin) {
  int tid = blockIdx.x * blockDim.x + threadIdx.x;
  int total = 3 * cin * HID_;
  if (tid >= total) return;
  int j    = tid & 15;
  int l    = (tid >> 4) & 31;
  int f    = (tid >> 9) & 31;
  int rest = tid >> 14;
  int nch  = cin >> 5;
  int c    = rest % nch;
  int k    = rest / nch;
  int K  = ((l >> 4) << 4) + j;
  int ci = c * 32 + K;
  int co = f * 16 + (l & 15);
  p[tid] = (_Float16)w[((size_t)k * cin + ci) * HID_ + co];
}

// ---------------------------------------------------------------------------
// Bilinear sampling: one wave per vertex, each lane handles 8 channels.
// Output X0 [B*N, 256] row-major f16.
// ---------------------------------------------------------------------------
__global__ void sample_kernel(const float* __restrict__ vert,
                              const float* __restrict__ feat,
                              _Float16* __restrict__ X0) {
  int tid  = blockIdx.x * 256 + threadIdx.x;
  int v    = tid >> 5;      // vertex 0..8191
  int lane = tid & 31;
  int b    = v >> 10;

  float y = vert[v * 2 + 0];
  float x = vert[v * 2 + 1];
  float y0f = floorf(y), x0f = floorf(x);
  float wy1 = y - y0f, wx1 = x - x0f;
  float wy0 = 1.f - wy1, wx0 = 1.f - wx1;
  int y0 = (int)y0f, x0 = (int)x0f;
  int c0 = lane * 8;

  float acc[8];
#pragma unroll
  for (int i = 0; i < 8; ++i) acc[i] = 0.f;

  int   ys[2]  = {y0, y0 + 1};
  int   xs[2]  = {x0, x0 + 1};
  float wys[2] = {wy0, wy1};
  float wxs[2] = {wx0, wx1};

#pragma unroll
  for (int iy = 0; iy < 2; ++iy) {
#pragma unroll
    for (int ix = 0; ix < 2; ++ix) {
      int yy = ys[iy], xx = xs[ix];
      float wgt = wys[iy] * wxs[ix];
      if (yy >= 0 && yy < H_ && xx >= 0 && xx < W_) {
        const float* p = feat + ((((size_t)b * H_ + yy) * W_ + xx) * C_ + c0);
        v4f p0 = *(const v4f*)p;
        v4f p1 = *(const v4f*)(p + 4);
#pragma unroll
        for (int i = 0; i < 4; ++i) {
          acc[i]     += wgt * p0[i];
          acc[4 + i] += wgt * p1[i];
        }
      }
    }
  }
  v8h out;
#pragma unroll
  for (int i = 0; i < 8; ++i) out[i] = (_Float16)acc[i];
  *(v8h*)(X0 + (size_t)v * C_ + c0) = out;
}

// ---------------------------------------------------------------------------
// Dilated K=3 Conv1D as WMMA GEMM.
// Block: 256 threads = 8 waves. Block tile: 32 rows x 512 couts.
// Wave tile: 32 x 64 (2 A-frags x 4 B-frags = 8 accum frags).
// Double-buffered LDS slab (50 rows x 32 ch), staged via async-to-LDS loads,
// shared across all 3 taps and all 8 waves per Cin-chunk.
// ---------------------------------------------------------------------------
template <int CIN>
__global__ void __launch_bounds__(256)
conv_wmma(const _Float16* __restrict__ X, const _Float16* __restrict__ Wp,
          const float* __restrict__ bias, _Float16* __restrict__ Y, int rate) {
  constexpr int NCH = CIN / 32;
  __shared__ __attribute__((aligned(16))) _Float16 slab[2][SLAB_ROWS * SLAB_STRIDE];

  const int tid   = threadIdx.x;
  const int lane  = tid & 31;
  const int wave  = tid >> 5;         // 0..7 -> co group of 64
  const int n0    = blockIdx.x * 32;
  const int bbase = n0 & ~(N_ - 1);   // batch start (N_ = 1024)
  const int hi    = lane >> 4;
  const int mrow  = lane & 15;

  // ---- slab loader role ----
  const int  row    = tid >> 2;       // 0..63 (only 0..49 used)
  const int  seg    = tid & 3;
  const bool loader = tid < 200;
  const int  grow   = n0 - MAXR + row;
  const int  lrow   = grow - bbase;
  const bool inb    = loader && (lrow >= 0) && (lrow < N_);

  const _Float16* gsrc = X + (size_t)grow * CIN + seg * 8;
  _Float16* ldst[2] = {&slab[0][row * SLAB_STRIDE + seg * 8],
                       &slab[1][row * SLAB_STRIDE + seg * 8]};

  // zero-fill halo rows outside the batch once (async loads never touch them)
  if (loader && !inb) {
    *(v8h*)ldst[0] = (v8h){};
    *(v8h*)ldst[1] = (v8h){};
  }

  auto issue = [&](int c, int buf) {
    if (inb) {
#if USE_ASYNC
      __builtin_amdgcn_global_load_async_to_lds_b128(
          (as1_v4i_p)(gsrc + c * 32), (as3_v4i_p)ldst[buf], 0, 0);
#else
      *(v8h*)ldst[buf] = *(const v8h*)(gsrc + c * 32);
#endif
    }
  };

  v8f acc[2][4] = {};

  issue(0, 0);
  for (int c = 0; c < NCH; ++c) {
    const int cur = c & 1;
    if (c + 1 < NCH) issue(c + 1, 1 - cur);
#if USE_ASYNC
    if (c + 1 < NCH) __builtin_amdgcn_s_wait_asynccnt(1);
    else             __builtin_amdgcn_s_wait_asynccnt(0);
#endif
    __syncthreads();
    const _Float16* sb = slab[cur];

#pragma unroll
    for (int k = 0; k < 3; ++k) {
      const int s = (k - 1) * rate;
      v16h a[2];
#pragma unroll
      for (int mi = 0; mi < 2; ++mi) {
        const int srow = mi * 16 + mrow + s + MAXR;   // 0..49 (rate <= 9)
        const _Float16* ap = sb + srow * SLAB_STRIDE + hi * 16;
        *(v8h*)&a[mi]       = *(const v8h*)(ap);       // K = hi*16 + 0..7
        *((v8h*)&a[mi] + 1) = *(const v8h*)(ap + 8);   // K = hi*16 + 8..15
      }
      const _Float16* bp =
          Wp + ((((size_t)k * NCH + c) * 32 + wave * 4) * 32 + lane) * 16;
#pragma unroll
      for (int f = 0; f < 4; ++f) {
        v16h bf;
        *(v8h*)&bf       = *(const v8h*)(bp + (size_t)f * 512 + 0);
        *((v8h*)&bf + 1) = *(const v8h*)(bp + (size_t)f * 512 + 8);
#pragma unroll
        for (int mi = 0; mi < 2; ++mi) {
          acc[mi][f] = __builtin_amdgcn_wmma_f32_16x16x32_f16(
              false, a[mi], false, bf, (short)0, acc[mi][f], false, false);
        }
      }
    }
    __syncthreads();
  }

  // epilogue: bias + ReLU + f16 store. D layout: lane -> N(col), vgpr -> M(row)
#pragma unroll
  for (int f = 0; f < 4; ++f) {
    const int   co = wave * 64 + f * 16 + mrow;
    const float bv = bias[co];
#pragma unroll
    for (int mi = 0; mi < 2; ++mi) {
#pragma unroll
      for (int vv = 0; vv < 8; ++vv) {
        const int rowo = n0 + mi * 16 + hi * 8 + vv;
        float val = acc[mi][f][vv] + bv;
        val = val > 0.f ? val : 0.f;
        Y[(size_t)rowo * HID_ + co] = (_Float16)val;
      }
    }
  }
}

// ---------------------------------------------------------------------------
// Offset head: 512 -> 2 GEMV per vertex (one wave each) + vertex add.
// ---------------------------------------------------------------------------
__global__ void offset_head(const _Float16* __restrict__ Hh,
                            const float* __restrict__ woff,
                            const float* __restrict__ vert,
                            float* __restrict__ out) {
  int tid  = blockIdx.x * 256 + threadIdx.x;
  int v    = tid >> 5;
  int lane = tid & 31;

  const _Float16* hp = Hh + (size_t)v * HID_ + lane * 16;
  v8h h0 = *(const v8h*)hp;
  v8h h1 = *(const v8h*)(hp + 8);

  const float* wp = woff + lane * 16 * 2;
  float s0 = 0.f, s1 = 0.f;
#pragma unroll
  for (int j = 0; j < 8; ++j) {
    float hv = (float)h0[j];
    s0 += hv * wp[2 * j];
    s1 += hv * wp[2 * j + 1];
  }
#pragma unroll
  for (int j = 0; j < 8; ++j) {
    float hv = (float)h1[j];
    s0 += hv * wp[16 + 2 * j];
    s1 += hv * wp[16 + 2 * j + 1];
  }
#pragma unroll
  for (int off = 16; off >= 1; off >>= 1) {
    s0 += __shfl_xor(s0, off, 32);
    s1 += __shfl_xor(s1, off, 32);
  }
  if (lane == 0) {
    out[v * 2 + 0] = vert[v * 2 + 0] + s0;
    out[v * 2 + 1] = vert[v * 2 + 1] + s1;
  }
}

// ---------------------------------------------------------------------------
extern "C" void kernel_launch(void* const* d_in, const int* in_sizes, int n_in,
                              void* d_out, int out_size, void* d_ws, size_t ws_size,
                              hipStream_t stream) {
  (void)in_sizes; (void)n_in; (void)out_size; (void)ws_size;

  const float* vertices = (const float*)d_in[0];
  const float* features = (const float*)d_in[1];
  const float* w[6];
  const float* bb[6];
  for (int i = 0; i < 6; ++i) {
    w[i]  = (const float*)d_in[2 + 2 * i];
    bb[i] = (const float*)d_in[3 + 2 * i];
  }
  const float* w_off = (const float*)d_in[14];
  float* out = (float*)d_out;

  char* ws = (char*)d_ws;
  _Float16* actA = (_Float16*)ws;                          // 8 MB
  _Float16* actB = (_Float16*)(ws + ((size_t)8 << 20));    // 8 MB
  _Float16* packs[6];
  size_t off = (size_t)16 << 20;
  const size_t sz0 = (size_t)3 * 256 * HID_ * 2;
  const size_t szi = (size_t)3 * 512 * HID_ * 2;
  packs[0] = (_Float16*)(ws + off); off += sz0;
  for (int i = 1; i < 6; ++i) { packs[i] = (_Float16*)(ws + off); off += szi; }

  // 1) pack weights to WMMA fragment order (f16)
  {
    int total0 = 3 * 256 * HID_;
    hipLaunchKernelGGL(pack_weights, dim3((total0 + 255) / 256), dim3(256), 0, stream,
                       w[0], packs[0], 256);
    int totali = 3 * 512 * HID_;
    for (int i = 1; i < 6; ++i)
      hipLaunchKernelGGL(pack_weights, dim3((totali + 255) / 256), dim3(256), 0, stream,
                         w[i], packs[i], 512);
  }

  // 2) bilinear gather -> actA [8192, 256] f16
  hipLaunchKernelGGL(sample_kernel, dim3((B_ * N_) / 8), dim3(256), 0, stream,
                     vertices, features, actA);

  // 3) conv stack (each block: 32 rows x full 512 couts)
  const int rates[6] = {1, 3, 9, 9, 3, 1};
  dim3 grid((B_ * N_) / 32), blk(256);
  hipLaunchKernelGGL((conv_wmma<256>), grid, blk, 0, stream,
                     actA, packs[0], bb[0], actB, rates[0]);
  _Float16* src = actB;
  _Float16* dst = actA;
  for (int i = 1; i < 6; ++i) {
    hipLaunchKernelGGL((conv_wmma<512>), grid, blk, 0, stream,
                       src, packs[i], bb[i], dst, rates[i]);
    _Float16* t = src; src = dst; dst = t;
  }

  // 4) offset head + vertex add -> d_out [B, N, 2] f32
  hipLaunchKernelGGL(offset_head, dim3((B_ * N_) / 8), dim3(256), 0, stream,
                     src, w_off, vertices, out);
}
